// LSTMModel_48661979463696
// MI455X (gfx1250) — compile-verified, matrix operand order checked by reading, not run on hardware
//
#include <hip/hip_runtime.h>

typedef __attribute__((ext_vector_type(16))) _Float16 v16h;
typedef __attribute__((ext_vector_type(8)))  float    v8f;

#define FIN 40
#define HID 64
#define TSTEPS 200
#define WPACK_BYTES 65536   // 16 tiles * 4 kchunks * 32 lanes * 16 f16

union V16U { uint4 u[2]; v16h v; };

__device__ __forceinline__ float sigf(float v) { return 1.0f / (1.0f + __expf(-v)); }

__device__ __forceinline__ float fast_tanh(float v) {
#if __has_builtin(__builtin_amdgcn_tanh_f32)
  return __builtin_amdgcn_tanh_f32(v);
#elif __has_builtin(__builtin_amdgcn_tanhf)
  return __builtin_amdgcn_tanhf(v);
#else
  return tanhf(v);
#endif
}

// ---------------------------------------------------------------------------
// Prep: pack [W_ih | 0 | W_hh] (f32) into the per-lane f16 B-fragment register
// image for v_wmma_f32_16x16x32_f16, tile order = gate-major (i,f,g,o) x 4
// column groups. Also fold b_ih + b_hh.
//   B-fragment (K=32 x N=16): lane (hf=lane>>4, n=lane&15) slot s holds
//   K = 8*hf + s          (s < 8)
//   K = 16 + 8*hf + (s-8) (s >= 8)
// Combined K axis: [0..39]=x proj, [40..63]=zero pad, [64..127]=h proj.
// ---------------------------------------------------------------------------
__global__ void lstm_pack(const float* __restrict__ W_ih, const float* __restrict__ W_hh,
                          const float* __restrict__ b_ih, const float* __restrict__ b_hh,
                          _Float16* __restrict__ wpack, float* __restrict__ bcomb) {
  int idx = blockIdx.x * blockDim.x + threadIdx.x;
  if (idx < 32768) {
    int tt = idx >> 11;          // tile 0..15  (G*4 + j)
    int kc = (idx >> 9) & 3;     // k-chunk 0..3
    int ln = (idx >> 4) & 31;    // lane
    int s  = idx & 15;           // slot within v16h
    int n  = ln & 15, hf = ln >> 4;
    int Kl = (s < 8) ? (8 * hf + s) : (16 + 8 * hf + (s - 8));
    int K  = kc * 32 + Kl;
    int G = tt >> 2, jj = tt & 3;
    int grow = G * 64 + jj * 16 + n;       // gate row in [0,256)
    float v = 0.f;
    if (K < FIN)      v = W_ih[grow * FIN + K];
    else if (K >= 64) v = W_hh[grow * HID + (K - 64)];
    wpack[idx] = (_Float16)v;
  }
  if (idx < 256) bcomb[idx] = b_ih[idx] + b_hh[idx];
}

// ---------------------------------------------------------------------------
// Fused LSTM: block = 4 waves = 16 batch rows. Wave j owns hidden columns
// [16j,16j+16): 16 WMMA/step, weight fragments register-resident, x loads
// software-pipelined one step ahead, h ping-pongs through 4KB LDS.
// ---------------------------------------------------------------------------
__global__ void __launch_bounds__(128) lstm_fused(
    const float* __restrict__ x, const _Float16* __restrict__ wpack,
    const float* __restrict__ bcomb, const float* __restrict__ W_fc,
    const float* __restrict__ b_fc, float* __restrict__ out) {
  __shared__ _Float16 sH[2][16 * HID];   // ping-pong hidden state (f16)

  const int tid  = threadIdx.x;
  const int lane = tid & 31;
  const int j    = tid >> 5;       // wave id = hidden column group
  const int hf   = lane >> 4;      // half-wave
  const int l16  = lane & 15;      // row (A) / column (B,C) index
  const int rowBase = blockIdx.x * 16;

  // ---- load this wave's 16 weight B-fragments into registers (loop-invariant)
  v16h wfrag[16];
  {
    const uint4* wg = (const uint4*)wpack;
    #pragma unroll
    for (int G = 0; G < 4; ++G) {
      #pragma unroll
      for (int kc = 0; kc < 4; ++kc) {
        const uint4* p = wg + ((size_t)((G * 4 + j) * 4 + kc) * 32 + lane) * 2;
        V16U u; u.u[0] = p[0]; u.u[1] = p[1];
        wfrag[G * 4 + kc] = u.v;
      }
    }
  }

  float bg[4];
  #pragma unroll
  for (int G = 0; G < 4; ++G) bg[G] = bcomb[G * 64 + j * 16 + l16];

  // zero h0 (buffer 0 only; buffer 1 is written before first read)
  ((uint4*)&sH[0][0])[tid] = make_uint4(0u, 0u, 0u, 0u);
  __syncthreads();

  const float* xrow = x + (size_t)(rowBase + l16) * (TSTEPS * FIN);

  float c[8];
  #pragma unroll
  for (int e = 0; e < 8; ++e) c[e] = 0.f;

  const float zsel = (hf == 0) ? 1.0f : 0.0f;   // branchless pad select

  // ---- software pipeline: preload x for t = 0 ----
  float4 p0 = *(const float4*)(xrow + 8 * hf);
  float4 p1 = *(const float4*)(xrow + 8 * hf + 4);
  float4 p2 = *(const float4*)(xrow + 16 + 8 * hf);
  float4 p3 = *(const float4*)(xrow + 16 + 8 * hf + 4);
  float4 p4 = *(const float4*)(xrow + 32);
  float4 p5 = *(const float4*)(xrow + 36);

  int cur = 0;
  for (int t = 0; t < TSTEPS; ++t) {
    // ---- A chunks 2,3 : K = 64..127 (h from LDS) — issue DS loads first ----
    V16U a2, a3;
    {
      const _Float16* hb = &sH[cur][0];
      a2.u[0] = *(const uint4*)(hb + l16 * HID + 8 * hf);
      a2.u[1] = *(const uint4*)(hb + l16 * HID + 16 + 8 * hf);
      a3.u[0] = *(const uint4*)(hb + l16 * HID + 32 + 8 * hf);
      a3.u[1] = *(const uint4*)(hb + l16 * HID + 48 + 8 * hf);
    }

    // ---- issue x loads for t+1 (clamped at the end; latency hidden by body)
    const int tn = (t + 1 < TSTEPS) ? (t + 1) : t;
    const float* xn = xrow + tn * FIN;
    float4 n0 = *(const float4*)(xn + 8 * hf);
    float4 n1 = *(const float4*)(xn + 8 * hf + 4);
    float4 n2 = *(const float4*)(xn + 16 + 8 * hf);
    float4 n3 = *(const float4*)(xn + 16 + 8 * hf + 4);
    float4 n4 = *(const float4*)(xn + 32);
    float4 n5 = *(const float4*)(xn + 36);
    __builtin_prefetch(xn + FIN, 0, 0);   // warm t+2

    // ---- build A chunks 0,1 from preloaded x (no vmem wait on this path) ----
    v16h A[4];
    {
      float t0[16] = {p0.x, p0.y, p0.z, p0.w, p1.x, p1.y, p1.z, p1.w,
                      p2.x, p2.y, p2.z, p2.w, p3.x, p3.y, p3.z, p3.w};
      #pragma unroll
      for (int i = 0; i < 16; ++i) A[0][i] = (_Float16)t0[i];
      float t1[8] = {p4.x, p4.y, p4.z, p4.w, p5.x, p5.y, p5.z, p5.w};
      #pragma unroll
      for (int i = 0; i < 8; ++i) A[1][i] = (_Float16)(t1[i] * zsel);
      #pragma unroll
      for (int i = 8; i < 16; ++i) A[1][i] = (_Float16)0.f;
    }
    A[2] = a2.v; A[3] = a3.v;

    v8f acc[4];
    #pragma unroll
    for (int G = 0; G < 4; ++G) {
      v8f a;
      #pragma unroll
      for (int e = 0; e < 8; ++e) a[e] = bg[G];
      #pragma unroll
      for (int kc = 0; kc < 4; ++kc) {
        a = __builtin_amdgcn_wmma_f32_16x16x32_f16(
                false, A[kc], false, wfrag[G * 4 + kc], (short)0, a, false, false);
      }
      acc[G] = a;
    }

    // ---- elementwise LSTM cell update (C-layout rows M = e + 8*hf) ----
    {
      _Float16* hn = &sH[cur ^ 1][0];
      #pragma unroll
      for (int e = 0; e < 8; ++e) {
        float iv = sigf(acc[0][e]);
        float fv = sigf(acc[1][e]);
        float gv = fast_tanh(acc[2][e]);
        float ov = sigf(acc[3][e]);
        c[e] = fv * c[e] + iv * gv;
        float hv = ov * fast_tanh(c[e]);
        hn[(e + 8 * hf) * HID + j * 16 + l16] = (_Float16)hv;
      }
    }

    // rotate x pipeline registers
    p0 = n0; p1 = n1; p2 = n2; p3 = n3; p4 = n4; p5 = n5;

    __syncthreads();   // new h visible to all waves for next step
    cur ^= 1;
  }

  // ---- final FC (H->1) + sigmoid on last hidden state ----
  if (tid < 16) {
    const _Float16* hb = &sH[cur][0];
    float s = b_fc[0];
    #pragma unroll
    for (int n = 0; n < HID; ++n) s += (float)hb[tid * HID + n] * W_fc[n];
    out[rowBase + tid] = sigf(s);
  }
}

// ---------------------------------------------------------------------------
extern "C" void kernel_launch(void* const* d_in, const int* in_sizes, int n_in,
                              void* d_out, int out_size, void* d_ws, size_t ws_size,
                              hipStream_t stream) {
  const float* x    = (const float*)d_in[0];
  const float* W_ih = (const float*)d_in[1];
  const float* W_hh = (const float*)d_in[2];
  const float* b_ih = (const float*)d_in[3];
  const float* b_hh = (const float*)d_in[4];
  const float* W_fc = (const float*)d_in[5];
  const float* b_fc = (const float*)d_in[6];
  float* out = (float*)d_out;

  _Float16* wpack = (_Float16*)d_ws;
  float*    bcomb = (float*)((char*)d_ws + WPACK_BYTES);

  lstm_pack<<<128, 256, 0, stream>>>(W_ih, W_hh, b_ih, b_hh, wpack, bcomb);

  int B = in_sizes[0] / (TSTEPS * FIN);   // 4096
  lstm_fused<<<B / 16, 128, 0, stream>>>(x, wpack, bcomb, W_fc, b_fc, out);
}
